// Generator_21930103013482
// MI455X (gfx1250) — compile-verified, matrix operand order checked by reading, not run on hardware
//
#include <hip/hip_runtime.h>
#include <hip/hip_bf16.h>
#include <math.h>

// ---------- types ----------
typedef __attribute__((ext_vector_type(16))) __bf16 v16bf;
typedef __attribute__((ext_vector_type(8)))  __bf16 v8bf;
typedef __attribute__((ext_vector_type(8)))  float  v8f;

#define HIDDEN 512
#define WINDOW 24
#define KERN   3
#define BATCH  2048
#define LCONV  (HIDDEN + KERN - 1)   // 514
#define K_EXT  (3 * HIDDEN)          // 1536 = [d0|s0|Hi]
#define HSTRIDE (WINDOW * HIDDEN)    // 12288, row stride of H per batch

// ---------- math helpers (branch-free) ----------
__device__ inline float fast_tanh(float x) {
  // tanh(x) = 1 - 2/(exp(2x)+1); saturates correctly at +/-inf
  float e = __expf(2.0f * x);
  return 1.0f - 2.0f / (e + 1.0f);
}
__device__ inline float fast_sigmoid(float x) {
  return 1.0f / (1.0f + __expf(-x));
}

// ---------- WMMA helpers ----------
__device__ inline v8f wmma_bf16(v16bf a, v16bf b, v8f c) {
  // D = A(16x32) * B(32x16) + C(16x16), f32 accumulate
  return __builtin_amdgcn_wmma_f32_16x16x32_bf16(
      /*neg_a=*/false, a, /*neg_b=*/false, b,
      /*c_mod=*/(short)0, c, /*reuse_a=*/false, /*reuse_b=*/false);
}

// Load a 16x32 bf16 fragment from a lane-resolved base pointer.
// Per ISA 16-bit A layout: lanes 0-15 -> rows r0..r0+15 with K = {0..7,16..23};
// lanes 16-31 -> same rows with K = {8..15,24..31}. The base pointer already
// includes row*ld + (lane-dependent +0/+8) so the loop adds only `k`.
__device__ inline v16bf load_frag_p(const __bf16* __restrict__ p) {
  v8bf lo = *(const v8bf*)(p);               // K = +0 .. +7
  v8bf hi = *(const v8bf*)(p + 16);          // K = +16 .. +23
  v16bf f;
#pragma unroll
  for (int i = 0; i < 8; ++i) { f[i] = lo[i]; f[i + 8] = hi[i]; }
  return f;
}

// lane-resolved base: row (r0 + lane%16), column offset (+8 for upper half-wave)
__device__ inline const __bf16* frag_base(const __bf16* base, int ld, int r0) {
  int lane = threadIdx.x & 31;
  return base + (size_t)(r0 + (lane & 15)) * ld + ((lane & 16) >> 1);
}

// ---------- conv: H[b][t][i] = relu(sum Z[c][b][i+k]*cw[t][c][k] + cb[t]) ----------
__global__ __launch_bounds__(256) void conv_kernel(
    const float* __restrict__ Z, const float* __restrict__ cw,
    const float* __restrict__ cb, __bf16* __restrict__ Hbf) {
  int idx = blockIdx.x * 256 + threadIdx.x;
  if (idx >= BATCH * WINDOW * HIDDEN) return;
  int i = idx & (HIDDEN - 1);
  int t = (idx >> 9) % WINDOW;
  int b = idx / (WINDOW * HIDDEN);
  float acc = cb[t];
#pragma unroll
  for (int c = 0; c < 2; ++c)
#pragma unroll
    for (int k = 0; k < KERN; ++k)
      acc += Z[(size_t)c * BATCH * LCONV + (size_t)b * LCONV + i + k] *
             cw[t * 6 + c * 3 + k];
  Hbf[idx] = (__bf16)fmaxf(acc, 0.0f);
}

// ---------- weight conversions ----------
__global__ __launch_bounds__(256) void cvt_wext_kernel(
    const float* __restrict__ Wd, const float* __restrict__ Ud,
    __bf16* __restrict__ Wext) {  // [512, 1536] = [Wd | Ud]
  int idx = blockIdx.x * 256 + threadIdx.x;
  if (idx >= HIDDEN * K_EXT) return;
  int n = idx / K_EXT, k = idx % K_EXT;
  float v = (k < 2 * HIDDEN) ? Wd[(size_t)n * (2 * HIDDEN) + k]
                             : Ud[(size_t)n * HIDDEN + (k - 2 * HIDDEN)];
  Wext[idx] = (__bf16)v;
}

__global__ __launch_bounds__(256) void cvt_whh_kernel(
    const float* __restrict__ Whh, __bf16* __restrict__ Whh_bf) {  // [2048,512]
  int idx = blockIdx.x * 256 + threadIdx.x;
  if (idx >= 4 * HIDDEN * HIDDEN) return;
  Whh_bf[idx] = (__bf16)Whh[idx];
}

// ---------- state init: d,s -> f32 copies + bf16 A_ds[:,0:512|512:1024] ----------
__global__ __launch_bounds__(256) void init_state_kernel(
    const float* __restrict__ d_in, const float* __restrict__ s_in,
    float* __restrict__ d0f, float* __restrict__ s0f,
    __bf16* __restrict__ Ads) {
  int idx = blockIdx.x * 256 + threadIdx.x;
  if (idx >= BATCH * HIDDEN) return;
  int b = idx >> 9, j = idx & (HIDDEN - 1);
  float dv = d_in[idx], sv = s_in[idx];
  d0f[idx] = dv; s0f[idx] = sv;
  Ads[(size_t)b * (2 * HIDDEN) + j] = (__bf16)dv;
  Ads[(size_t)b * (2 * HIDDEN) + HIDDEN + j] = (__bf16)sv;
}

// ---------- yr[b] = y_real[b,:] . w[0:24] + b0 ----------
__global__ __launch_bounds__(256) void yr_kernel(
    const float* __restrict__ y_real, const float* __restrict__ w,
    const float* __restrict__ b0, float* __restrict__ yr) {
  int b = blockIdx.x * 256 + threadIdx.x;
  if (b >= BATCH) return;
  float acc = b0[0];
#pragma unroll
  for (int j = 0; j < WINDOW; ++j) acc += y_real[b * WINDOW + j] * w[j];
  yr[b] = acc;
}

// Macro for the double-buffered GEMM step: prefetch k-next, multiply current.
#define GEMM_STEP(PA0, PA1, PB0, PB1, OFF)                    \
  {                                                           \
    v16bf na0 = load_frag_p((PA0) + (OFF));                   \
    v16bf na1 = load_frag_p((PA1) + (OFF));                   \
    v16bf nb0 = load_frag_p((PB0) + (OFF##_B));               \
    v16bf nb1 = load_frag_p((PB1) + (OFF##_B));               \
    acc[0][0] = wmma_bf16(a0, b0, acc[0][0]);                 \
    acc[0][1] = wmma_bf16(a0, b1, acc[0][1]);                 \
    acc[1][0] = wmma_bf16(a1, b0, acc[1][0]);                 \
    acc[1][1] = wmma_bf16(a1, b1, acc[1][1]);                 \
    a0 = na0; a1 = na1; b0 = nb0; b1 = nb1;                   \
  }

// ---------- GEMM1: l[b] += sum_n tanh(([d0|s0|Hi] @ Wext^T)[b,n]) * vd[n] ----------
// Two-phase, double-buffered K loop with hoisted lane base pointers:
// phase 1 (k<1024) reads Ads (stride 1024), phase 2 reads Hbf_t (stride 12288).
// Constant strides per loop -> pure pointer+immediate addressing in the body.
__global__ __launch_bounds__(256) void gemm1_kernel(
    const __bf16* __restrict__ Ads,     // [2048,1024], K slice 0..1023
    const __bf16* __restrict__ Hbf_t,   // H + t*512, row stride HSTRIDE, K 1024..1535
    const __bf16* __restrict__ Wext,    // [512,1536]
    const float* __restrict__ vd, float* __restrict__ l) {
  int wave = threadIdx.x >> 5;
  int wm = wave >> 1, wn = wave & 1;
  int m0 = blockIdx.y * 128 + wm * 32;
  int n0 = blockIdx.x * 64 + wn * 32;

  const __bf16* pa0 = frag_base(Ads, 2 * HIDDEN, m0);
  const __bf16* pa1 = frag_base(Ads, 2 * HIDDEN, m0 + 16);
  const __bf16* ph0 = frag_base(Hbf_t, HSTRIDE, m0);
  const __bf16* ph1 = frag_base(Hbf_t, HSTRIDE, m0 + 16);
  const __bf16* pb0 = frag_base(Wext, K_EXT, n0);
  const __bf16* pb1 = frag_base(Wext, K_EXT, n0 + 16);

  v8f acc[2][2] = {};
  v16bf a0 = load_frag_p(pa0);
  v16bf a1 = load_frag_p(pa1);
  v16bf b0 = load_frag_p(pb0);
  v16bf b1 = load_frag_p(pb1);

  // phase 1: k = 32 .. 1023 (A from Ads)
  for (int k = 32; k < 2 * HIDDEN; k += 32) {
    int k_B = k;
    GEMM_STEP(pa0, pa1, pb0, pb1, k)
  }
  // boundary: prefetch first Hi chunk (global k = 1024)
  {
    int z = 0, z_B = 2 * HIDDEN;
    GEMM_STEP(ph0, ph1, pb0, pb1, z)
  }
  // phase 2: kk = 32 .. 511 (A from Hbf_t; B at 1024+kk)
  for (int kk = 32; kk < HIDDEN; kk += 32) {
    int kk_B = 2 * HIDDEN + kk;
    GEMM_STEP(ph0, ph1, pb0, pb1, kk)
  }
  // drain
  acc[0][0] = wmma_bf16(a0, b0, acc[0][0]);
  acc[0][1] = wmma_bf16(a0, b1, acc[0][1]);
  acc[1][0] = wmma_bf16(a1, b0, acc[1][0]);
  acc[1][1] = wmma_bf16(a1, b1, acc[1][1]);

  // epilogue: tanh * vd, reduce over N within each 16-lane half, atomicAdd into l
  int lane = threadIdx.x & 31;
  int nlo = lane & 15;
  int mhalf = (lane >> 4) << 3;  // 0 or 8
  float v0 = vd[n0 + nlo], v1 = vd[n0 + 16 + nlo];
#pragma unroll
  for (int mi = 0; mi < 2; ++mi) {
    float rs[8];
#pragma unroll
    for (int v = 0; v < 8; ++v) {
      float r = fast_tanh(acc[mi][0][v]) * v0 + fast_tanh(acc[mi][1][v]) * v1;
      r += __shfl_xor(r, 1, 16);
      r += __shfl_xor(r, 2, 16);
      r += __shfl_xor(r, 4, 16);
      r += __shfl_xor(r, 8, 16);
      rs[v] = r;
    }
    if (nlo == 0) {
      int m = m0 + mi * 16 + mhalf;
#pragma unroll
      for (int v = 0; v < 8; ++v) atomicAdd(&l[m + v], rs[v]);
    }
  }
}

// ---------- softmax over batch ----------
__global__ __launch_bounds__(256) void softmax_kernel(
    const float* __restrict__ l, float* __restrict__ Bw) {
  __shared__ float red[256];
  int tid = threadIdx.x;
  float m = -INFINITY;
  for (int i = tid; i < BATCH; i += 256) m = fmaxf(m, l[i]);
  red[tid] = m; __syncthreads();
  for (int s = 128; s > 0; s >>= 1) {
    if (tid < s) red[tid] = fmaxf(red[tid], red[tid + s]);
    __syncthreads();
  }
  m = red[0]; __syncthreads();
  float sum = 0.f;
  for (int i = tid; i < BATCH; i += 256) sum += __expf(l[i] - m);
  red[tid] = sum; __syncthreads();
  for (int s = 128; s > 0; s >>= 1) {
    if (tid < s) red[tid] += red[tid + s];
    __syncthreads();
  }
  float inv = 1.0f / red[0];
  for (int i = tid; i < BATCH; i += 256) Bw[i] = __expf(l[i] - m) * inv;
}

// ---------- ct_vec += Bw @ Hi  (parallel over batch slices) ----------
__global__ __launch_bounds__(512) void ct_reduce_kernel(
    const float* __restrict__ Bw, const __bf16* __restrict__ Hbf_t,
    float* __restrict__ ct_vec) {
  __shared__ float sBw[64];
  int tid = threadIdx.x;
  int b0 = blockIdx.x * 64;
  if (tid < 64) sBw[tid] = Bw[b0 + tid];
  __syncthreads();
  float acc = 0.f;
#pragma unroll 4
  for (int i = 0; i < 64; ++i)
    acc += sBw[i] * (float)Hbf_t[(size_t)(b0 + i) * HSTRIDE + tid];
  atomicAdd(&ct_vec[tid], acc);
}

// ---------- scal = ct_vec . w[24:] ----------
__global__ __launch_bounds__(512) void ct_dot_kernel(
    const float* __restrict__ ct_vec, const float* __restrict__ w,
    float* __restrict__ scal) {
  __shared__ float red[512];
  int tid = threadIdx.x;
  red[tid] = ct_vec[tid] * w[WINDOW + tid];
  __syncthreads();
  for (int s = 256; s > 0; s >>= 1) {
    if (tid < s) red[tid] += red[tid + s];
    __syncthreads();
  }
  if (tid == 0) scal[0] = red[0];
}

// ---------- GEMM2: gates_mm = d0 @ Whh^T  ([2048,512] x [512,2048]) ----------
__global__ __launch_bounds__(256) void gemm2_kernel(
    const __bf16* __restrict__ Ads,     // d0 half, row stride 1024
    const __bf16* __restrict__ Whh_bf,  // [2048,512] row-major
    float* __restrict__ gates) {        // [2048,2048]
  int wave = threadIdx.x >> 5;
  int wm = wave >> 1, wn = wave & 1;
  int m0 = blockIdx.y * 128 + wm * 32;
  int n0 = blockIdx.x * 64 + wn * 32;

  const __bf16* pa0 = frag_base(Ads, 2 * HIDDEN, m0);
  const __bf16* pa1 = frag_base(Ads, 2 * HIDDEN, m0 + 16);
  const __bf16* pb0 = frag_base(Whh_bf, HIDDEN, n0);
  const __bf16* pb1 = frag_base(Whh_bf, HIDDEN, n0 + 16);

  v8f acc[2][2] = {};
  v16bf a0 = load_frag_p(pa0);
  v16bf a1 = load_frag_p(pa1);
  v16bf b0 = load_frag_p(pb0);
  v16bf b1 = load_frag_p(pb1);
  for (int k = 32; k < HIDDEN; k += 32) {
    int k_B = k;
    GEMM_STEP(pa0, pa1, pb0, pb1, k)
  }
  acc[0][0] = wmma_bf16(a0, b0, acc[0][0]);
  acc[0][1] = wmma_bf16(a0, b1, acc[0][1]);
  acc[1][0] = wmma_bf16(a1, b0, acc[1][0]);
  acc[1][1] = wmma_bf16(a1, b1, acc[1][1]);

  int lane = threadIdx.x & 31;
  int nlo = lane & 15;
  int mhalf = (lane >> 4) << 3;
#pragma unroll
  for (int mi = 0; mi < 2; ++mi)
#pragma unroll
    for (int ni = 0; ni < 2; ++ni) {
      int n = n0 + ni * 16 + nlo;
      int m = m0 + mi * 16 + mhalf;
#pragma unroll
      for (int v = 0; v < 8; ++v)
        gates[(size_t)(m + v) * (4 * HIDDEN) + n] = acc[mi][ni][v];
    }
}

// ---------- pointwise LSTM cell (adds rank-1 y_fake*Wih + biases) ----------
__global__ __launch_bounds__(256) void lstm_kernel(
    const float* __restrict__ gates, const float* __restrict__ yr,
    const float* __restrict__ scal, const float* __restrict__ Wih,
    const float* __restrict__ bih, const float* __restrict__ bhh,
    float* __restrict__ d0f, float* __restrict__ s0f,
    __bf16* __restrict__ Ads, float* __restrict__ yf) {
  int idx = blockIdx.x * 256 + threadIdx.x;
  if (idx >= BATCH * HIDDEN) return;
  int b = idx >> 9, j = idx & (HIDDEN - 1);
  float y = yr[b] + scal[0];
  const float* g = gates + (size_t)b * (4 * HIDDEN);
  float gi = g[j]              + y * Wih[j]              + bih[j]              + bhh[j];
  float gf = g[HIDDEN + j]     + y * Wih[HIDDEN + j]     + bih[HIDDEN + j]     + bhh[HIDDEN + j];
  float gg = g[2 * HIDDEN + j] + y * Wih[2 * HIDDEN + j] + bih[2 * HIDDEN + j] + bhh[2 * HIDDEN + j];
  float go = g[3 * HIDDEN + j] + y * Wih[3 * HIDDEN + j] + bih[3 * HIDDEN + j] + bhh[3 * HIDDEN + j];
  float si = fast_sigmoid(gi);
  float sf = fast_sigmoid(gf);
  float so = fast_sigmoid(go);
  float s = sf * s0f[idx] + si * fast_tanh(gg);
  float d = so * fast_tanh(s);
  s0f[idx] = s; d0f[idx] = d;
  Ads[(size_t)b * (2 * HIDDEN) + j] = (__bf16)d;
  Ads[(size_t)b * (2 * HIDDEN) + HIDDEN + j] = (__bf16)s;
  if (j == 0) yf[b] = y;
}

// ---------- pack outputs: [y_fake(2048) | d0(1M) | s0(1M)] ----------
__global__ __launch_bounds__(256) void finalize_kernel(
    const float* __restrict__ yf, const float* __restrict__ d0f,
    const float* __restrict__ s0f, float* __restrict__ out) {
  int idx = blockIdx.x * 256 + threadIdx.x;
  const int n1 = BATCH, n2 = BATCH + BATCH * HIDDEN, n3 = BATCH + 2 * BATCH * HIDDEN;
  if (idx >= n3) return;
  if (idx < n1)      out[idx] = yf[idx];
  else if (idx < n2) out[idx] = d0f[idx - n1];
  else               out[idx] = s0f[idx - n2];
}

// ---------- host ----------
static inline char* carve(char*& p, size_t bytes) {
  char* r = p;
  p += (bytes + 255) & ~(size_t)255;
  return r;
}

extern "C" void kernel_launch(void* const* d_in, const int* in_sizes, int n_in,
                              void* d_out, int out_size, void* d_ws, size_t ws_size,
                              hipStream_t stream) {
  const float* Z      = (const float*)d_in[0];
  const float* d_st   = (const float*)d_in[1];
  const float* s_st   = (const float*)d_in[2];
  const float* y_real = (const float*)d_in[3];
  const float* vd     = (const float*)d_in[4];
  const float* Wd     = (const float*)d_in[5];
  const float* Ud     = (const float*)d_in[6];
  const float* w      = (const float*)d_in[7];
  const float* b0     = (const float*)d_in[8];
  const float* conv_w = (const float*)d_in[9];
  const float* conv_b = (const float*)d_in[10];
  const float* Wih    = (const float*)d_in[11];
  const float* Whh    = (const float*)d_in[12];
  const float* bih    = (const float*)d_in[13];
  const float* bhh    = (const float*)d_in[14];
  float* out = (float*)d_out;

  char* p = (char*)d_ws;
  __bf16* Hbf    = (__bf16*)carve(p, (size_t)BATCH * WINDOW * HIDDEN * 2);   // 50.3 MB
  __bf16* Ads    = (__bf16*)carve(p, (size_t)BATCH * 2 * HIDDEN * 2);        // 4.2 MB
  __bf16* Wext   = (__bf16*)carve(p, (size_t)HIDDEN * K_EXT * 2);            // 1.6 MB
  __bf16* Whh_bf = (__bf16*)carve(p, (size_t)4 * HIDDEN * HIDDEN * 2);       // 2.1 MB
  float*  gates  = (float*)carve(p, (size_t)BATCH * 4 * HIDDEN * 4);         // 16.8 MB
  float*  d0f    = (float*)carve(p, (size_t)BATCH * HIDDEN * 4);
  float*  s0f    = (float*)carve(p, (size_t)BATCH * HIDDEN * 4);
  float*  lbuf   = (float*)carve(p, (size_t)BATCH * 4);
  float*  Bw     = (float*)carve(p, (size_t)BATCH * 4);
  float*  yr     = (float*)carve(p, (size_t)BATCH * 4);
  float*  yf     = (float*)carve(p, (size_t)BATCH * 4);
  float*  ct_vec = (float*)carve(p, (size_t)HIDDEN * 4);
  float*  scal   = (float*)carve(p, 256);

  // ---- init ----
  conv_kernel<<<(BATCH * WINDOW * HIDDEN + 255) / 256, 256, 0, stream>>>(
      Z, conv_w, conv_b, Hbf);
  cvt_wext_kernel<<<(HIDDEN * K_EXT + 255) / 256, 256, 0, stream>>>(Wd, Ud, Wext);
  cvt_whh_kernel<<<(4 * HIDDEN * HIDDEN + 255) / 256, 256, 0, stream>>>(Whh, Whh_bf);
  init_state_kernel<<<(BATCH * HIDDEN + 255) / 256, 256, 0, stream>>>(
      d_st, s_st, d0f, s0f, Ads);
  yr_kernel<<<(BATCH + 255) / 256, 256, 0, stream>>>(y_real, w, b0, yr);
  hipMemsetAsync(ct_vec, 0, HIDDEN * sizeof(float), stream);

  // ---- 24 sequential steps ----
  for (int t = 0; t < WINDOW; ++t) {
    const __bf16* Hbf_t = Hbf + (size_t)t * HIDDEN;
    hipMemsetAsync(lbuf, 0, BATCH * sizeof(float), stream);
    // GEMM1: [2048,1536] x [1536,512]^T, fused tanh * vd row-reduce
    gemm1_kernel<<<dim3(HIDDEN / 64, BATCH / 128), 256, 0, stream>>>(
        Ads, Hbf_t, Wext, vd, lbuf);
    softmax_kernel<<<1, 256, 0, stream>>>(lbuf, Bw);
    ct_reduce_kernel<<<BATCH / 64, 512, 0, stream>>>(Bw, Hbf_t, ct_vec);
    ct_dot_kernel<<<1, 512, 0, stream>>>(ct_vec, w, scal);
    // GEMM2: d0 @ Whh^T -> gates matmul part
    gemm2_kernel<<<dim3(4 * HIDDEN / 64, BATCH / 128), 256, 0, stream>>>(
        Ads, Whh_bf, gates);
    lstm_kernel<<<(BATCH * HIDDEN + 255) / 256, 256, 0, stream>>>(
        gates, yr, scal, Wih, bih, bhh, d0f, s0f, Ads, yf);
  }

  finalize_kernel<<<(BATCH + 2 * BATCH * HIDDEN + 255) / 256, 256, 0, stream>>>(
      yf, d0f, s0f, out);
}
